// CTAttention_49761491091652
// MI455X (gfx1250) — matrix operand order repaired and checked by reading, not compiled, not run on hardware
//
#include <hip/hip_runtime.h>
#include <hip/hip_bf16.h>

// ---------------- problem constants ----------------
#define KW     24                 // window size
#define DIL    4                  // dilation
#define CDIM   256                // channels
#define NH     8                  // heads
#define HDIM   32                 // head dim
#define NTOK   500000             // tokens
#define BATCH  8                  // batch size (pad fill id)
#define BLOCKL (KW * DIL)         // 96
#define NAP    500064             // padded token count (multiple of 96 and 32)
#define NWIN   (NAP / KW)         // 20836 windows
#define ATT_SCALE 0.17677669529663687f   // 32^-0.5
#define MASKV  (-1000.0f)

typedef __attribute__((ext_vector_type(16))) _Float16 v16h;
typedef __attribute__((ext_vector_type(8)))  _Float16 v8h;
typedef __attribute__((ext_vector_type(8)))  float    v8f;

// ---- CDNA5 async global->LDS copy (ASYNCcnt path, ISA 10.x/15.18) ----
// GVS mode: mem = SGPR64 base + VGPR32 byte offset. VDST VGPR = LDS byte addr.
__device__ __forceinline__ void async_ld128(const void* gbase, unsigned byte_off,
                                            void* lds_dst) {
  unsigned lds_off = (unsigned)(uintptr_t)lds_dst;   // LDS aperture: low 32 bits
  asm volatile("global_load_async_to_lds_b128 %0, %1, %2"
               :: "v"(lds_off), "v"(byte_off), "s"(gbase)
               : "memory");
}
__device__ __forceinline__ void wait_asynccnt0() {
  asm volatile("s_wait_asynccnt 0x0" ::: "memory");
}

// ---- WMMA operand loaders (wave32, f16 16x16x32) ----
// A operand (16x32): lane m=L&15 is row m; chunks [hi*8..+7] and [16+hi*8..+7].
__device__ __forceinline__ v16h load_a16(const _Float16* base, int stride) {
  const int lane = threadIdx.x & 31;
  const int m = lane & 15, hi = lane >> 4;
  const _Float16* row = base + m * stride;
  v8h lo  = *(const v8h*)(row + hi * 8);
  v8h hi8 = *(const v8h*)(row + 16 + hi * 8);
  v16h a;
#pragma unroll
  for (int t = 0; t < 8; ++t) { a[t] = lo[t]; a[t + 8] = hi8[t]; }
  return a;
}
// B operand (32x16): lane n=L&15 is column n; chunks [hi*8..+7] and [16+hi*8..+7].
__device__ __forceinline__ v16h load_b16(const _Float16* base, int stride) {
  const int lane = threadIdx.x & 31;
  const int n = lane & 15, hi = lane >> 4;
  const _Float16* row = base + n * stride;
  v8h lo  = *(const v8h*)(row + hi * 8);
  v8h hi8 = *(const v8h*)(row + 16 + hi * 8);
  v16h b;
#pragma unroll
  for (int t = 0; t < 8; ++t) { b[t] = lo[t]; b[t + 8] = hi8[t]; }
  return b;
}
__device__ __forceinline__ v8f wmma_f16(v16h a, v16h b, v8f c) {
  return __builtin_amdgcn_wmma_f32_16x16x32_f16(false, a, false, b,
                                                (short)0, c, false, false);
}

// ---------------- kernel 1: fused pad + QKV GEMM ----------------
// Block: 256 thr = 8 waves; tile 32 rows x 256 cols; wave owns 32x32 (2x2 WMMA
// tiles, shared fragments -> 1 DS fragment load per WMMA). K=256 staged f16.
#define GS 264   // LDS row stride in halves (256 + 8 -> bank spread)
__global__ __launch_bounds__(256) void qkv_gemm(
    const float* __restrict__ X, const float* __restrict__ W,
    const float* __restrict__ bias, _Float16* __restrict__ out) {
  __shared__ __align__(16) _Float16 As[32 * GS];    // ~16.9 KB
  __shared__ __align__(16) _Float16 Bs[256 * GS];   // ~132 KB
  const int rowBase = blockIdx.x * 32;
  const int colBase = blockIdx.y * 256;
  const int t = threadIdx.x;
  // stage A (32x256 f32 -> f16), rows >= NTOK are zero (padding)
#pragma unroll
  for (int it = 0; it < 8; ++it) {
    int lin = t + it * 256;                 // float4 index
    int r = lin >> 6, c4 = (lin & 63) << 2;
    int gr = rowBase + r;
    float4 f = (gr < NTOK) ? *(const float4*)(X + (size_t)gr * CDIM + c4)
                           : make_float4(0.f, 0.f, 0.f, 0.f);
    _Float16* dst = As + r * GS + c4;
    dst[0] = (_Float16)f.x; dst[1] = (_Float16)f.y;
    dst[2] = (_Float16)f.z; dst[3] = (_Float16)f.w;
  }
  // stage B (256x256 f32 -> f16): weight output channels (L2-resident)
#pragma unroll 4
  for (int it = 0; it < 64; ++it) {
    int lin = t + it * 256;
    int r = lin >> 6, c4 = (lin & 63) << 2;
    float4 f = *(const float4*)(W + (size_t)(colBase + r) * CDIM + c4);
    _Float16* dst = Bs + r * GS + c4;
    dst[0] = (_Float16)f.x; dst[1] = (_Float16)f.y;
    dst[2] = (_Float16)f.z; dst[3] = (_Float16)f.w;
  }
  __syncthreads();
  const int wv = t >> 5;                    // wave = 32-col strip
  const int lane = t & 31;
  const int n = lane & 15, hi = lane >> 4;
  v8f acc00 = {}, acc01 = {}, acc10 = {}, acc11 = {};
#pragma unroll
  for (int k0 = 0; k0 < 256; k0 += 32) {
    v16h a0 = load_a16(As + k0, GS);
    v16h a1 = load_a16(As + 16 * GS + k0, GS);
    v16h b0 = load_b16(Bs + (wv * 32) * GS + k0, GS);
    v16h b1 = load_b16(Bs + (wv * 32 + 16) * GS + k0, GS);
    acc00 = wmma_f16(a0, b0, acc00);
    acc01 = wmma_f16(a0, b1, acc01);
    acc10 = wmma_f16(a1, b0, acc10);
    acc11 = wmma_f16(a1, b1, acc11);
  }
  const float bv0 = bias[colBase + wv * 32 + n];
  const float bv1 = bias[colBase + wv * 32 + 16 + n];
  __syncthreads();                          // everyone done reading As/Bs
  // repack D (f16) into LDS (reuse As) for coalesced global stores
  _Float16* Ds = As;                        // 32 rows x 256 cols, stride GS
#pragma unroll
  for (int r = 0; r < 8; ++r) {
    int r0 = r + hi * 8;
    Ds[r0 * GS + wv * 32 + n]             = (_Float16)(acc00[r] + bv0);
    Ds[r0 * GS + wv * 32 + 16 + n]        = (_Float16)(acc01[r] + bv1);
    Ds[(16 + r0) * GS + wv * 32 + n]      = (_Float16)(acc10[r] + bv0);
    Ds[(16 + r0) * GS + wv * 32 + 16 + n] = (_Float16)(acc11[r] + bv1);
  }
  __syncthreads();
#pragma unroll
  for (int it = 0; it < 4; ++it) {          // 32 rows x 32 v8h chunks
    int lin = t + it * 256;
    int r = lin >> 5, c8 = (lin & 31) << 3;
    *(v8h*)(out + (size_t)(rowBase + r) * 768 + colBase + c8) =
        *(const v8h*)(Ds + r * GS + c8);
  }
}

// ---------------- kernel 2: dilated window attention ----------------
// One block per window, one wave per head. Row map: p(w,i)=(w/4)*96+i*4+(w&3).
#define TS 40    // attention tile stride in halves (32 + 8 pad)
__global__ __launch_bounds__(256) void win_attn(
    const _Float16* __restrict__ qkv, const int* __restrict__ bidx,
    _Float16* __restrict__ aout) {
  __shared__ __align__(16) _Float16 Qs[NH][32 * TS];
  __shared__ __align__(16) _Float16 Ks[NH][32 * TS];
  __shared__ __align__(16) _Float16 Vt[NH][32 * TS];   // transposed V: Vt[d][j]
  __shared__ float    Ss[NH][32 * 33];                 // f32 scores
  __shared__ __align__(16) _Float16 Ps[NH][32 * TS];   // softmax probs (f16)
  __shared__ int bi[32];
  const int w = blockIdx.x;
  const int t = threadIdx.x;
  const int h = t >> 5;
  const int lane = t & 31;
  const int pbase = (w >> 2) * BLOCKL + (w & 3);
  if (t < 32) {
    int p = pbase + t * DIL;
    bi[t] = (t < KW && p < NTOK) ? bidx[p] : BATCH;
  }
  // stage q/k via async global->LDS; v via b128 regs -> transposed DS stores
  {
    const int i = lane;
    _Float16* qd = &Qs[h][i * TS];
    _Float16* kd = &Ks[h][i * TS];
    if (i < KW) {
      unsigned go = (unsigned)((((unsigned)(pbase + i * DIL)) * 768u
                               + (unsigned)(h * HDIM)) * 2u);
#pragma unroll
      for (int u = 0; u < 4; ++u) {
        async_ld128(qkv, go + u * 16, qd + u * 8);          // q
        async_ld128(qkv, go + 512 + u * 16, kd + u * 8);    // k (+256 halves)
      }
      const _Float16* vrow =
          qkv + (size_t)(pbase + i * DIL) * 768 + h * HDIM + 512;
      v8h vv[4];
#pragma unroll
      for (int u = 0; u < 4; ++u) vv[u] = *(const v8h*)(vrow + u * 8);
#pragma unroll
      for (int u = 0; u < 4; ++u)
#pragma unroll
        for (int e = 0; e < 8; ++e) Vt[h][(u * 8 + e) * TS + i] = vv[u][e];
    } else {
#pragma unroll
      for (int u = 0; u < HDIM; ++u) { qd[u] = (_Float16)0.f; kd[u] = (_Float16)0.f; }
#pragma unroll
      for (int d = 0; d < HDIM; ++d) Vt[h][d * TS + i] = (_Float16)0.f;
    }
  }
  wait_asynccnt0();
  __syncthreads();
  const int n = lane & 15, hi = lane >> 4;
  // S = (q @ k^T) * scale + mask   (2x2 WMMA tiles, K=32)
#pragma unroll
  for (int mt = 0; mt < 2; ++mt)
#pragma unroll
    for (int nt = 0; nt < 2; ++nt) {
      v8f acc = {};
      v16h a = load_a16(&Qs[h][mt * 16 * TS], TS);
      v16h b = load_b16(&Ks[h][nt * 16 * TS], TS);
      acc = wmma_f16(a, b, acc);
      const int col = nt * 16 + n;
#pragma unroll
      for (int r = 0; r < 8; ++r) {
        int row = mt * 16 + r + hi * 8;
        float val = acc[r] * ATT_SCALE + ((bi[row] != bi[col]) ? MASKV : 0.f);
        if (col >= KW) val = -1e30f;   // kill 24->32 column padding
        Ss[h][row * 33 + col] = val;
      }
    }
  __syncthreads();
  // softmax over 24 valid columns, re-stage as f16 P (K-pad zeroed)
  {
    _Float16* pd = &Ps[h][lane * TS];
    if (lane < KW) {
      float vals[KW];
      float mx = -1e30f;
#pragma unroll
      for (int j = 0; j < KW; ++j) { vals[j] = Ss[h][lane * 33 + j]; mx = fmaxf(mx, vals[j]); }
      float s = 0.f;
#pragma unroll
      for (int j = 0; j < KW; ++j) { vals[j] = __expf(vals[j] - mx); s += vals[j]; }
      float inv = 1.f / s;
#pragma unroll
      for (int j = 0; j < KW; ++j) pd[j] = (_Float16)(vals[j] * inv);
#pragma unroll
      for (int j = KW; j < 32; ++j) pd[j] = (_Float16)0.f;
    } else {
#pragma unroll
      for (int j = 0; j < 32; ++j) pd[j] = (_Float16)0.f;
    }
  }
  __syncthreads();
  // O = P @ V, scatter-store to un-reshuffled rows (reverse dilation is free)
#pragma unroll
  for (int mt = 0; mt < 2; ++mt)
#pragma unroll
    for (int nt = 0; nt < 2; ++nt) {
      v8f acc = {};
      v16h a = load_a16(&Ps[h][mt * 16 * TS], TS);
      v16h b = load_b16(&Vt[h][nt * 16 * TS], TS);
      acc = wmma_f16(a, b, acc);
      const int d = nt * 16 + n;
#pragma unroll
      for (int r = 0; r < 8; ++r) {
        int row = mt * 16 + r + hi * 8;
        if (row < KW) {
          size_t p = (size_t)pbase + (size_t)row * DIL;
          aout[p * CDIM + h * HDIM + d] = (_Float16)acc[r];
        }
      }
    }
}

// ---------------- kernel 3: output projection ----------------
// Block: 256 thr = 8 waves; tile 32 rows x 256 cols (full width); A staged
// via async global->LDS b128 (already f16); f32 output repacked for float4 stores.
__global__ __launch_bounds__(256) void proj_gemm(
    const _Float16* __restrict__ Xh, const float* __restrict__ W,
    const float* __restrict__ bias, float* __restrict__ out) {
  __shared__ __align__(16) _Float16 As[32 * GS];
  __shared__ __align__(16) _Float16 Bs[256 * GS];
  const int rowBase = blockIdx.x * 32;
  const int t = threadIdx.x;
  // stage A: 32x256 halves = 1024 x 16B async copies
#pragma unroll
  for (int it = 0; it < 4; ++it) {
    int lin = t + it * 256;
    int r = lin >> 5, c8 = (lin & 31) << 3;
    unsigned go = (unsigned)((((unsigned)(rowBase + r)) * CDIM + c8) * 2u);
    async_ld128(Xh, go, As + r * GS + c8);
  }
  // stage B: full proj_w 256x256 f32 -> f16 (L2-resident)
#pragma unroll 4
  for (int it = 0; it < 64; ++it) {
    int lin = t + it * 256;
    int r = lin >> 6, c4 = (lin & 63) << 2;
    float4 f = *(const float4*)(W + (size_t)r * CDIM + c4);
    _Float16* dst = Bs + r * GS + c4;
    dst[0] = (_Float16)f.x; dst[1] = (_Float16)f.y;
    dst[2] = (_Float16)f.z; dst[3] = (_Float16)f.w;
  }
  wait_asynccnt0();
  __syncthreads();
  const int wv = t >> 5;
  const int lane = t & 31;
  const int n = lane & 15, hi = lane >> 4;
  v8f acc00 = {}, acc01 = {}, acc10 = {}, acc11 = {};
#pragma unroll
  for (int k0 = 0; k0 < 256; k0 += 32) {
    v16h a0 = load_a16(As + k0, GS);
    v16h a1 = load_a16(As + 16 * GS + k0, GS);
    v16h b0 = load_b16(Bs + (wv * 32) * GS + k0, GS);
    v16h b1 = load_b16(Bs + (wv * 32 + 16) * GS + k0, GS);
    acc00 = wmma_f16(a0, b0, acc00);
    acc01 = wmma_f16(a0, b1, acc01);
    acc10 = wmma_f16(a1, b0, acc10);
    acc11 = wmma_f16(a1, b1, acc11);
  }
  const float bv0 = bias[wv * 32 + n];
  const float bv1 = bias[wv * 32 + 16 + n];
  __syncthreads();
  // repack f32 D into LDS (reuse Bs) for coalesced float4 stores
  float* Df = (float*)Bs;                   // 32 rows x 256 cols, stride 260
#pragma unroll
  for (int r = 0; r < 8; ++r) {
    int r0 = r + hi * 8;
    Df[r0 * 260 + wv * 32 + n]             = acc00[r] + bv0;
    Df[r0 * 260 + wv * 32 + 16 + n]        = acc01[r] + bv1;
    Df[(16 + r0) * 260 + wv * 32 + n]      = acc10[r] + bv0;
    Df[(16 + r0) * 260 + wv * 32 + 16 + n] = acc11[r] + bv1;
  }
  __syncthreads();
#pragma unroll
  for (int it = 0; it < 8; ++it) {          // 32 rows x 64 float4 chunks
    int lin = t + it * 256;
    int r = lin >> 6, c4 = (lin & 63) << 2;
    *(float4*)(out + (size_t)(rowBase + r) * CDIM + c4) =
        *(const float4*)(Df + r * 260 + c4);
  }
}

extern "C" void kernel_launch(void* const* d_in, const int* in_sizes, int n_in,
                              void* d_out, int out_size, void* d_ws, size_t ws_size,
                              hipStream_t stream) {
  const float* data   = (const float*)d_in[0];
  const float* qkv_w  = (const float*)d_in[1];
  const float* qkv_b  = (const float*)d_in[2];
  const float* proj_w = (const float*)d_in[3];
  const float* proj_b = (const float*)d_in[4];
  const int*   bidx   = (const int*)d_in[5];
  float* out = (float*)d_out;

  // workspace: QKV f16 [NAP,768] (768 MB) + attn-out f16 [NAP,256] (256 MB)
  _Float16* qkv  = (_Float16*)d_ws;
  _Float16* aout = qkv + (size_t)NAP * 768;

  qkv_gemm <<<dim3(NAP / 32, 768 / 256), 256, 0, stream>>>(data, qkv_w, qkv_b, qkv);
  win_attn <<<dim3(NWIN), 256, 0, stream>>>(qkv, bidx, aout);
  proj_gemm<<<dim3(NTOK / 32), 256, 0, stream>>>(aout, proj_w, proj_b, out);
}
